// GroupPointEncoder_49529562857912
// MI455X (gfx1250) — matrix operand order repaired and checked by reading, not compiled, not run on hardware
//
#include <hip/hip_runtime.h>

typedef _Float16 half16 __attribute__((ext_vector_type(16)));
typedef float    f32x8  __attribute__((ext_vector_type(8)));

#define BDIM     8
#define NDIM     2048
#define K1       384     // 3*NUM_FEATS
#define N1       512     // 4*NUM_FEATS
#define N2       256     // 2*NUM_FEATS
#define TM       64      // rows per block
#define THREADS  256
#define MTOT     65536   // G*B*N

union H16 { half16 h; unsigned u[8]; _Float16 e[16]; };

__global__ __launch_bounds__(THREADS)
void fused_group_point_encoder(const float* __restrict__ point_coord,
                               const int*   __restrict__ labels,
                               const float* __restrict__ pc_range,
                               const float* __restrict__ noise,
                               const float* __restrict__ query_pos,
                               const float* __restrict__ label_weight,
                               const float* __restrict__ W1,
                               const float* __restrict__ b1,
                               const float* __restrict__ W2,
                               const float* __restrict__ b2,
                               float* __restrict__ out)
{
    // 64 KiB LDS: first holds pe tile (64x384 f16 = 48 KiB), then reused for h tile (64x512 f16)
    __shared__ __align__(16) unsigned char smem_raw[TM * N1 * 2];
    _Float16* sh_f16 = (_Float16*)smem_raw;
    unsigned* sh_u32 = (unsigned*)smem_raw;

    const int tid     = threadIdx.x;
    const int lane    = tid & 31;
    const int wid     = tid >> 5;      // wave 0..7
    const int hlf     = lane >> 4;     // 0/1: lane half
    const int nlo     = lane & 15;
    const int rowbase = blockIdx.x * TM;

    // ---------------- phase 0a: copy query_pos -> out[..., 0:256] (coalesced f4) ----
    {
        const float4* qp4  = (const float4*)query_pos;
        float4*       out4 = (float4*)out;
        #pragma unroll
        for (int i = 0; i < 16; ++i) {
            int idx = tid + i * THREADS;     // 0..4095 over 64 rows x 64 float4
            int r   = idx >> 6;
            int c   = idx & 63;
            int gid = rowbase + r;
            out4[gid * 128 + c] = qp4[gid * 64 + c];
        }
    }

    // ---------------- phase 0b: sine embedding tile into LDS (f16) -------------------
    {
        const int r0  = tid >> 2;                 // 0..63 : row in tile
        const int sub = tid & 3;                  // 4 threads per row
        const int gid = rowbase + r0;
        const int rem = gid & (BDIM * NDIM - 1);  // b*N + n
        const int g   = gid >> 14;                // B*N = 16384 = 2^14
        const int lab = labels[rem];
        const float stdv  = (lab == 0 || lab >= 6) ? 2.0f : 4.0f;
        const float scale = (g == 0) ? 0.0f : stdv;   // group 0 uses clean coords

        float nv[3];
        #pragma unroll
        for (int d = 0; d < 3; ++d) {
            float lo = pc_range[d];
            float hi = pc_range[3 + d];
            float c  = point_coord[rem * 3 + d] + noise[gid * 3 + d] * scale;
            nv[d] = (c - lo) / (hi - lo) * 6.2831853071795864f;  // * 2*pi
        }
        const float NEG_L2T_64 = -0.20762050593046f;  // -log2(10000)/64
        #pragma unroll 4
        for (int i = 0; i < 96; ++i) {
            int f = sub * 96 + i;
            int d = (f < 128) ? 1 : ((f < 256) ? 0 : 2);   // concat order: y, x, z
            int k = f & 127;
            float v = nv[d] * exp2f((float)(k >> 1) * NEG_L2T_64);
            float e = (k & 1) ? __cosf(v) : __sinf(v);
            sh_f16[r0 * K1 + f] = (_Float16)e;
        }
    }
    __syncthreads();

    // ---------------- GEMM1: h = relu(pe @ W1^T + b1), tile 64x512, K=384 ------------
    f32x8 acc1[4][4];
    {
        const f32x8 z = {0.f,0.f,0.f,0.f,0.f,0.f,0.f,0.f};
        #pragma unroll
        for (int rt = 0; rt < 4; ++rt)
            #pragma unroll
            for (int ct = 0; ct < 4; ++ct) acc1[rt][ct] = z;
    }
    const int nb1 = wid * 64;                     // this wave's column base in N1
    const float2* W1f2 = (const float2*)W1;

    #pragma unroll 1
    for (int kk = 0; kk < K1; kk += 32) {
        const int kh = kk >> 1;
        half16 a[4];
        #pragma unroll
        for (int rt = 0; rt < 4; ++rt) {
            // A 16x32 f16 layout: VGPR j holds K = (j>>2)*16 + hlf*8 + (j&3)*2 (pair)
            H16 t;
            const int base = (rt * 16 + nlo) * (K1 / 2) + kh + hlf * 4;
            #pragma unroll
            for (int j = 0; j < 8; ++j)
                t.u[j] = sh_u32[base + (j >> 2) * 8 + (j & 3)];
            a[rt] = t.h;
        }
        #pragma unroll
        for (int ct = 0; ct < 4; ++ct) {
            // B 32x16 f16 layout: VGPR j holds K = (j>>2)*16 + hlf*8 + (j&3)*2 (pair)
            const int n = nb1 + ct * 16 + nlo;
            H16 t;
            const int base = n * (K1 / 2) + kh + hlf * 4;
            #pragma unroll
            for (int j = 0; j < 8; ++j) {
                float2 w = W1f2[base + (j >> 2) * 8 + (j & 3)];
                t.e[2 * j]     = (_Float16)w.x;
                t.e[2 * j + 1] = (_Float16)w.y;
            }
            #pragma unroll
            for (int rt = 0; rt < 4; ++rt)
                acc1[rt][ct] = __builtin_amdgcn_wmma_f32_16x16x32_f16(
                    false, a[rt], false, t.h, (short)0, acc1[rt][ct], false, false);
        }
    }

    // bias + relu -> h tile (f16) back into LDS (reuses pe region)
    float bias1[4];
    #pragma unroll
    for (int ct = 0; ct < 4; ++ct) bias1[ct] = b1[nb1 + ct * 16 + nlo];

    __syncthreads();   // everyone done reading pe
    #pragma unroll
    for (int rt = 0; rt < 4; ++rt)
        #pragma unroll
        for (int ct = 0; ct < 4; ++ct) {
            const int n = nb1 + ct * 16 + nlo;
            #pragma unroll
            for (int r = 0; r < 8; ++r) {
                const int m = rt * 16 + r + hlf * 8;   // C/D layout: VGPR r -> M=r (+8 for hi half)
                float v = acc1[rt][ct][r] + bias1[ct];
                sh_f16[m * N1 + n] = (_Float16)(v > 0.f ? v : 0.f);
            }
        }
    __syncthreads();

    // ---------------- GEMM2: pos_emb = h @ W2^T + b2, tile 64x256, K=512 -------------
    f32x8 acc2[4][2];
    {
        const f32x8 z = {0.f,0.f,0.f,0.f,0.f,0.f,0.f,0.f};
        #pragma unroll
        for (int rt = 0; rt < 4; ++rt) { acc2[rt][0] = z; acc2[rt][1] = z; }
    }
    const int nb2 = wid * 32;
    const float2* W2f2 = (const float2*)W2;

    #pragma unroll 1
    for (int kk = 0; kk < N1; kk += 32) {
        const int kh = kk >> 1;
        half16 a[4];
        #pragma unroll
        for (int rt = 0; rt < 4; ++rt) {
            H16 t;
            const int base = (rt * 16 + nlo) * (N1 / 2) + kh + hlf * 4;
            #pragma unroll
            for (int j = 0; j < 8; ++j)
                t.u[j] = sh_u32[base + (j >> 2) * 8 + (j & 3)];
            a[rt] = t.h;
        }
        #pragma unroll
        for (int ct = 0; ct < 2; ++ct) {
            const int n = nb2 + ct * 16 + nlo;
            H16 t;
            const int base = n * (N1 / 2) + kh + hlf * 4;
            #pragma unroll
            for (int j = 0; j < 8; ++j) {
                float2 w = W2f2[base + (j >> 2) * 8 + (j & 3)];
                t.e[2 * j]     = (_Float16)w.x;
                t.e[2 * j + 1] = (_Float16)w.y;
            }
            #pragma unroll
            for (int rt = 0; rt < 4; ++rt)
                acc2[rt][ct] = __builtin_amdgcn_wmma_f32_16x16x32_f16(
                    false, a[rt], false, t.h, (short)0, acc2[rt][ct], false, false);
        }
    }

    // ---------------- epilogue: + b2 + label_weight[labels], write out[..., 256:512] --
    float bias2[2];
    bias2[0] = b2[nb2 + nlo];
    bias2[1] = b2[nb2 + 16 + nlo];

    #pragma unroll
    for (int rt = 0; rt < 4; ++rt) {
        #pragma unroll
        for (int r = 0; r < 8; ++r) {
            const int m   = rt * 16 + r + hlf * 8;
            const int gid = rowbase + m;
            const int lab = labels[gid & (BDIM * NDIM - 1)];
            #pragma unroll
            for (int ct = 0; ct < 2; ++ct) {
                const int n = nb2 + ct * 16 + nlo;
                float q = acc2[rt][ct][r] + bias2[ct] + label_weight[lab * N2 + n];
                out[gid * 512 + 256 + n] = q;
            }
        }
    }
}

extern "C" void kernel_launch(void* const* d_in, const int* in_sizes, int n_in,
                              void* d_out, int out_size, void* d_ws, size_t ws_size,
                              hipStream_t stream) {
    const float* point_coord  = (const float*)d_in[0];
    const int*   labels       = (const int*)  d_in[1];
    const float* pc_range     = (const float*)d_in[2];
    const float* noise        = (const float*)d_in[3];
    const float* query_pos    = (const float*)d_in[4];
    const float* label_weight = (const float*)d_in[5];
    const float* W1           = (const float*)d_in[6];
    const float* b1           = (const float*)d_in[7];
    const float* W2           = (const float*)d_in[8];
    const float* b2           = (const float*)d_in[9];
    float* out = (float*)d_out;

    dim3 grid(MTOT / TM);   // 1024 blocks x 64 rows
    dim3 block(THREADS);    // 8 wave32 waves
    hipLaunchKernelGGL(fused_group_point_encoder, grid, block, 0, stream,
                       point_coord, labels, pc_range, noise, query_pos,
                       label_weight, W1, b1, W2, b2, out);
}
